// SqrllLayer_52621939311017
// MI455X (gfx1250) — matrix operand order repaired and checked by reading, not compiled, not run on hardware
//
#include <hip/hip_runtime.h>
#include <hip/hip_bf16.h>

typedef __bf16 bf16_t;
typedef __attribute__((ext_vector_type(16))) __bf16 v16bf;
typedef __attribute__((ext_vector_type(8)))  float  v8f;
typedef __attribute__((ext_vector_type(4)))  unsigned int v4u;

struct Bits256 { v4u lo; v4u hi; };
static_assert(sizeof(Bits256) == sizeof(v16bf), "frag size mismatch");

__device__ __forceinline__ v16bf make_frag(v4u lo, v4u hi) {
    Bits256 b;
    b.lo = lo; b.hi = hi;
    return __builtin_bit_cast(v16bf, b);
}

// CDNA5 async global->LDS copy (ASYNCcnt-tracked, no VGPR staging).
// VDST VGPR carries the wave-relative LDS byte offset; for generic LDS
// pointers the low 32 bits ARE that offset (flat-aperture rule, ISA 10.2).
__device__ __forceinline__ void async_load_b128(const bf16_t* g, bf16_t* l) {
    unsigned lds_off = (unsigned)(size_t)l;
    asm volatile("global_load_async_to_lds_b128 %0, %1, off"
                 :: "v"(lds_off), "v"(g) : "memory");
}
// Async loads complete IN ORDER (ISA 10.7.1): waiting for <= N outstanding
// proves every load issued before the newest N has completed.
__device__ __forceinline__ void wait_async_le0() {
    asm volatile("s_wait_asynccnt 0" ::: "memory");
}
__device__ __forceinline__ void wait_async_le6() {
    asm volatile("s_wait_asynccnt 6" ::: "memory");
}

// ---------------------------------------------------------------------------
// f32 -> bf16 conversion (grid-stride)
// ---------------------------------------------------------------------------
__global__ void cvt_bf16_kernel(const float* __restrict__ src,
                                bf16_t* __restrict__ dst, long n) {
    long i = (long)blockIdx.x * blockDim.x + threadIdx.x;
    long stride = (long)gridDim.x * blockDim.x;
    for (; i < n; i += stride) dst[i] = (bf16_t)src[i];
}

// ---------------------------------------------------------------------------
// Tiled bf16 WMMA GEMM:  out[M,N] = A[M,K] * W[N,K]^T  (+ fused epilogue)
//   mode 0: out = acc                     (f32 store)
//   mode 1: out = sigmoid(acc + bias[n])  (f32 store)
//   mode 2: out = acc * mulsrc[m*N+n]     (f32 store)
// Block tile 256x128, 8 waves (4 M x 2 N), wave tile 64x64, K-step 32.
// Triple-buffered LDS, 2-tile-deep global_load_async_to_lds_b128 pipeline.
// ---------------------------------------------------------------------------
#define BM 256
#define BN 128
#define BK 32
#define LDSTR 40  // LDS row stride in bf16 elements (80B, 16B-aligned rows)

__global__ __launch_bounds__(256)
void gemm_bf16_kernel(const bf16_t* __restrict__ A, const bf16_t* __restrict__ W,
                      const float* __restrict__ bias, const float* __restrict__ mulsrc,
                      float* __restrict__ out, int M, int N, int K, int mode)
{
    __shared__ __attribute__((aligned(16))) bf16_t lA[3][BM * LDSTR];
    __shared__ __attribute__((aligned(16))) bf16_t lB[3][BN * LDSTR];

    const int tid  = threadIdx.x;
    const int lane = tid & 31;
    const int wid  = tid >> 5;
    const int wm   = (wid & 3) * 64;   // wave M offset in block tile
    const int wn   = (wid >> 2) * 64;  // wave N offset in block tile
    const int half = lane >> 4;        // 0: lanes 0-15, 1: lanes 16-31
    const int l15  = lane & 15;

    const int mBase = blockIdx.y * BM;
    const int nBase = blockIdx.x * BN;

    // Stage a 256x32 tile of A (1024 x 16B) and a 128x32 tile of W (512 x 16B)
    // straight into LDS via async DMA; 256 threads -> 4 + 2 chunks each.
    auto loadTile = [&](int kb, int buf) {
        #pragma unroll
        for (int c = 0; c < 4; ++c) {
            int i   = tid + c * 256;
            int row = i >> 2;          // 4 x 16B chunks per 64B row
            int col = (i & 3) * 8;     // element offset within row
            async_load_b128(A + (size_t)(mBase + row) * K + kb + col,
                            &lA[buf][row * LDSTR + col]);
        }
        #pragma unroll
        for (int c = 0; c < 2; ++c) {
            int i   = tid + c * 256;
            int row = i >> 2;
            int col = (i & 3) * 8;
            async_load_b128(W + (size_t)(nBase + row) * K + kb + col,
                            &lB[buf][row * LDSTR + col]);
        }
    };

    v8f zero = {0.f, 0.f, 0.f, 0.f, 0.f, 0.f, 0.f, 0.f};
    v8f acc[4][4];
    #pragma unroll
    for (int i = 0; i < 4; ++i)
        #pragma unroll
        for (int j = 0; j < 4; ++j) acc[i][j] = zero;

    const int nK = K / BK;   // 64 for K=2048

    loadTile(0, 0);
    if (nK > 1) loadTile(BK, 1);
    wait_async_le6();        // in-order completion => tile 0 resident
    __syncthreads();

    int bc = 0;              // buffer being computed from
    for (int kt = 0; kt < nK; ++kt) {
        if (kt + 2 < nK) {
            int bl = bc + 2; if (bl >= 3) bl -= 3;
            loadTile((kt + 2) * BK, bl);
        }

        // A fragments (16x32 bf16 per 16-row sub-tile):
        // lane<16 holds K {0..7,16..23}; lane>=16 holds K {8..15,24..31}.
        v16bf af[4];
        #pragma unroll
        for (int i = 0; i < 4; ++i) {
            const bf16_t* base = &lA[bc][(wm + i * 16 + l15) * LDSTR + half * 8];
            v4u lo = *(const v4u*)base;          // K = half*8 + 0..7   -> vgpr 0..3
            v4u hi = *(const v4u*)(base + 16);   // K = 16+half*8+0..7  -> vgpr 4..7
            af[i] = make_frag(lo, hi);
        }
        // B fragments (32x16 bf16): lane<16 -> K {0..7,16..23}; lane>=16 -> K {8..15,24..31}.
        v16bf bfrag[4];
        #pragma unroll
        for (int j = 0; j < 4; ++j) {
            const bf16_t* base = &lB[bc][(wn + j * 16 + l15) * LDSTR + half * 8];
            v4u lo = *(const v4u*)base;          // K = half*8 + 0..7   -> vgpr 0..3
            v4u hi = *(const v4u*)(base + 16);   // K = 16+half*8+0..7  -> vgpr 4..7
            bfrag[j] = make_frag(lo, hi);
        }

        #pragma unroll
        for (int i = 0; i < 4; ++i)
            #pragma unroll
            for (int j = 0; j < 4; ++j)
                acc[i][j] = __builtin_amdgcn_wmma_f32_16x16x32_bf16(
                    false, af[i], false, bfrag[j], (short)0, acc[i][j], false, false);

        // Guarantee tile kt+1 is resident before the buffer-swap barrier:
        // with tile kt+2's 6 loads just issued, <=6 outstanding means all
        // older (tile kt+1) loads finished.  On the tail, drain fully.
        if (kt + 2 < nK) wait_async_le6();
        else             wait_async_le0();
        __syncthreads();

        bc = (bc + 1 == 3) ? 0 : bc + 1;
    }

    // Epilogue. C/D layout: vgpr v -> M = v + 8*half, N = lane&15.
    #pragma unroll
    for (int i = 0; i < 4; ++i) {
        #pragma unroll
        for (int j = 0; j < 4; ++j) {
            const int n = nBase + wn + j * 16 + l15;
            #pragma unroll
            for (int v = 0; v < 8; ++v) {
                const int m = mBase + wm + i * 16 + half * 8 + v;
                float val = acc[i][j][v];
                if (mode == 1) {
                    val = 1.0f / (1.0f + __expf(-(val + bias[n])));
                } else if (mode == 2) {
                    val = val * mulsrc[(size_t)m * N + n];
                }
                out[(size_t)m * N + n] = val;
            }
        }
    }
}

// ---------------------------------------------------------------------------
// Sequential recurrence per (batch, channel):  h_t = r_t * h_{t-1} + xi_t
// xi is overwritten with h.  Final h written to d_out tail (mem_out).
// ---------------------------------------------------------------------------
__global__ void sqrll_scan_kernel(const float* __restrict__ r,
                                  float* __restrict__ xi,   // becomes h
                                  const float* __restrict__ mem0,
                                  float* __restrict__ mem_out,
                                  int B, int T, int C)
{
    int idx = blockIdx.x * blockDim.x + threadIdx.x;   // 0 .. B*C-1
    if (idx >= B * C) return;
    int b = idx / C;
    int c = idx - b * C;

    float h = mem0[(size_t)b * C + c];
    const float* rp = r  + (size_t)b * T * C + c;
    float*       xp = xi + (size_t)b * T * C + c;
    #pragma unroll 8
    for (int t = 0; t < T; ++t) {
        float rr = rp[(size_t)t * C];
        float xx = xp[(size_t)t * C];
        h = rr * h + xx;
        xp[(size_t)t * C] = h;
    }
    mem_out[(size_t)b * C + c] = h;
}

// ---------------------------------------------------------------------------
// y = softsign(h) * og, packed to bf16 for the final GEMM.
// ---------------------------------------------------------------------------
__global__ void softsign_gate_kernel(const float* __restrict__ h,
                                     const float* __restrict__ og,
                                     bf16_t* __restrict__ yb, long n)
{
    long i = (long)blockIdx.x * blockDim.x + threadIdx.x;
    long stride = (long)gridDim.x * blockDim.x;
    for (; i < n; i += stride) {
        float hv = h[i];
        float yv = hv / (1.0f + fabsf(hv)) * og[i];
        yb[i] = (bf16_t)yv;
    }
}

// ---------------------------------------------------------------------------
extern "C" void kernel_launch(void* const* d_in, const int* in_sizes, int n_in,
                              void* d_out, int out_size, void* d_ws, size_t ws_size,
                              hipStream_t stream) {
    (void)in_sizes; (void)n_in; (void)out_size; (void)ws_size;

    const float* x     = (const float*)d_in[0];
    const float* mem   = (const float*)d_in[1];
    const float* wr_w  = (const float*)d_in[2];
    const float* wr_b  = (const float*)d_in[3];
    const float* wi_w  = (const float*)d_in[4];
    const float* wig_w = (const float*)d_in[5];
    const float* wig_b = (const float*)d_in[6];
    const float* wog_w = (const float*)d_in[7];
    const float* wog_b = (const float*)d_in[8];
    const float* wo_w  = (const float*)d_in[9];
    float* out = (float*)d_out;

    const int Bb = 4, T = 2048, C = 2048;
    const long MT = (long)Bb * T;        // 8192 rows
    const long NX = MT * C;              // 16,777,216 activations
    const long NW = (long)C * C;         // 4,194,304 per weight

    // ---- workspace carve-up ----
    char* ws = (char*)d_ws;
    size_t off = 0;
    bf16_t* xb   = (bf16_t*)(ws + off); off += (size_t)NX * 2;   // x in bf16
    bf16_t* wrb  = (bf16_t*)(ws + off); off += (size_t)NW * 2;
    bf16_t* wib  = (bf16_t*)(ws + off); off += (size_t)NW * 2;
    bf16_t* wigb = (bf16_t*)(ws + off); off += (size_t)NW * 2;
    bf16_t* wogb = (bf16_t*)(ws + off); off += (size_t)NW * 2;
    bf16_t* wob  = (bf16_t*)(ws + off); off += (size_t)NW * 2;
    float*  ig   = (float*)(ws + off);  off += (size_t)NX * 4;
    float*  og   = (float*)(ws + off);  off += (size_t)NX * 4;
    float*  rb   = (float*)(ws + off);  off += (size_t)NX * 4;
    float*  xi   = (float*)(ws + off);  off += (size_t)NX * 4;   // becomes h
    bf16_t* yb   = (bf16_t*)(ws + off); off += (size_t)NX * 2;

    // ---- f32 -> bf16 conversions ----
    cvt_bf16_kernel<<<4096, 256, 0, stream>>>(x, xb, NX);
    cvt_bf16_kernel<<<2048, 256, 0, stream>>>(wr_w,  wrb,  NW);
    cvt_bf16_kernel<<<2048, 256, 0, stream>>>(wi_w,  wib,  NW);
    cvt_bf16_kernel<<<2048, 256, 0, stream>>>(wig_w, wigb, NW);
    cvt_bf16_kernel<<<2048, 256, 0, stream>>>(wog_w, wogb, NW);
    cvt_bf16_kernel<<<2048, 256, 0, stream>>>(wo_w,  wob,  NW);

    // ---- four input GEMMs with fused epilogues ----
    dim3 grid(C / BN, (unsigned)(MT / BM));   // (16, 32)
    gemm_bf16_kernel<<<grid, 256, 0, stream>>>(xb, wigb, wig_b, nullptr, ig,
                                               (int)MT, C, C, /*sigmoid*/1);
    gemm_bf16_kernel<<<grid, 256, 0, stream>>>(xb, wogb, wog_b, nullptr, og,
                                               (int)MT, C, C, /*sigmoid*/1);
    gemm_bf16_kernel<<<grid, 256, 0, stream>>>(xb, wrb,  wr_b,  nullptr, rb,
                                               (int)MT, C, C, /*sigmoid*/1);
    gemm_bf16_kernel<<<grid, 256, 0, stream>>>(xb, wib,  nullptr, ig,    xi,
                                               (int)MT, C, C, /*mul ig*/2);

    // ---- recurrence (h overwrites xi; mem_out to output tail) ----
    sqrll_scan_kernel<<<(Bb * C + 255) / 256, 256, 0, stream>>>(
        rb, xi, mem, out + NX, Bb, T, C);

    // ---- softsign * og -> bf16 ----
    softsign_gate_kernel<<<4096, 256, 0, stream>>>(xi, og, yb, NX);

    // ---- output projection ----
    gemm_bf16_kernel<<<grid, 256, 0, stream>>>(yb, wob, nullptr, nullptr, out,
                                               (int)MT, C, C, /*plain*/0);
}
